// GRU_18107582120193
// MI455X (gfx1250) — matrix operand order repaired and checked by reading, not compiled, not run on hardware
//
#include <hip/hip_runtime.h>
#include <hip/hip_bf16.h>

// ---------------------------------------------------------------------------
// GRU on MI455X (gfx1250): persistent wave32 WMMA scan kernel.
//   N_IN=256, N_H=1024, N_OUT=64, BATCH=64, T=512
//   Recurrent GEMM fused with input projection: A=[h|u_t] (K=1280),
//   B = [W_h ; W_u]^T pre-converted to bf16, v_wmma_f32_16x16x32_bf16.
// ---------------------------------------------------------------------------

#define N_IN   256
#define N_H    1024
#define N_OUT  64
#define BATCH  64
#define TSTEPS 512
#define KCAT   (N_H + N_IN)      // 1280
#define NWG    64                // workgroups in persistent kernel
#define BLOCK  256               // 8 wave32 waves per WG -> 512 waves total

typedef __attribute__((ext_vector_type(16))) __bf16 v16bf;
typedef __attribute__((ext_vector_type(8)))  __bf16 v8bf;
typedef __attribute__((ext_vector_type(8)))  float  v8f;

// ------------------------------- setup kernels -----------------------------

__global__ void k_cvt_u(const float* __restrict__ u, __bf16* __restrict__ ubf, int n) {
    int i = blockIdx.x * blockDim.x + threadIdx.x;
    if (i < n) ubf[i] = (__bf16)u[i];
}

// WT[n][k] (n<1024, k<1280) = k<1024 ? W_h[k][n] : W_u[k-1024][n], in bf16.
__global__ void k_wt_gate(const float* __restrict__ Wh, const float* __restrict__ Wu,
                          __bf16* __restrict__ WT) {
    int i = blockIdx.x * blockDim.x + threadIdx.x;
    if (i >= N_H * KCAT) return;
    int n = i / KCAT, k = i % KCAT;
    float v = (k < N_H) ? Wh[k * N_H + n] : Wu[(k - N_H) * N_H + n];
    WT[i] = (__bf16)v;
}

// WTy[n][k] (n<64, k<1024) = W_hy[k][n] in bf16.
__global__ void k_wt_y(const float* __restrict__ Why, __bf16* __restrict__ WTy) {
    int i = blockIdx.x * blockDim.x + threadIdx.x;
    if (i >= N_OUT * N_H) return;
    int n = i / N_H, k = i % N_H;
    WTy[i] = (__bf16)Why[k * N_OUT + n];
}

__global__ void k_init_h(const float* __restrict__ h0, float* __restrict__ h,
                         __bf16* __restrict__ hbf, unsigned* __restrict__ bar) {
    int i = blockIdx.x * blockDim.x + threadIdx.x;
    if (i < BATCH * N_H) {
        float v = h0[i & (N_H - 1)];
        h[i] = v;
        hbf[i] = (__bf16)v;
    }
    if (i < 2) bar[i] = 0u;   // arrival count, generation
}

// ----------------------------- device helpers ------------------------------

__device__ __forceinline__ void grid_barrier(unsigned* cnt, unsigned* gen) {
    __threadfence();
    __syncthreads();
    if (threadIdx.x == 0) {
        unsigned old  = __hip_atomic_load(gen, __ATOMIC_RELAXED, __HIP_MEMORY_SCOPE_AGENT);
        unsigned prev = __hip_atomic_fetch_add(cnt, 1u, __ATOMIC_ACQ_REL, __HIP_MEMORY_SCOPE_AGENT);
        if (prev == NWG - 1) {
            __hip_atomic_store(cnt, 0u, __ATOMIC_RELAXED, __HIP_MEMORY_SCOPE_AGENT);
            __hip_atomic_fetch_add(gen, 1u, __ATOMIC_RELEASE, __HIP_MEMORY_SCOPE_AGENT);
        } else {
            while (__hip_atomic_load(gen, __ATOMIC_ACQUIRE, __HIP_MEMORY_SCOPE_AGENT) == old) {
                __builtin_amdgcn_s_sleep(1);
            }
        }
    }
    __syncthreads();
    __threadfence();
}

// One 16x16 output tile, K = 1280 = [A1row (1024) | Urow (256)], B = WTn (col-major-K).
// halfSel = (lane>=16). A layout: lanes0-15 K 0..15, lanes16-31 K 16..31.
// B layout: lanes0-15 K 0..15, lanes16-31 K 16..31.
__device__ __forceinline__ v8f gemm_tile_kcat(const __bf16* __restrict__ Arow,
                                              const __bf16* __restrict__ Urow,
                                              const __bf16* __restrict__ WTn,
                                              int halfSel) {
    v8f acc = {};
#pragma unroll
    for (int kc = 0; kc < KCAT / 32; ++kc) {
        const int k0 = kc * 32;
        const __bf16* asrc = (k0 < N_H) ? (Arow + k0) : (Urow + (k0 - N_H));
        const __bf16* ap = asrc + halfSel * 16;
        v16bf a = *(const v16bf*)(ap);
        v16bf b = *(const v16bf*)(WTn + k0 + halfSel * 16);
        acc = __builtin_amdgcn_wmma_f32_16x16x32_bf16(false, a, false, b,
                                                      (short)0, acc, false, false);
    }
    return acc;
}

// One 16x16 output tile, K = 1024 (h only), for the y projection.
__device__ __forceinline__ v8f gemm_tile_h(const __bf16* __restrict__ Arow,
                                           const __bf16* __restrict__ WTn,
                                           int halfSel) {
    v8f acc = {};
#pragma unroll
    for (int kc = 0; kc < N_H / 32; ++kc) {
        const int k0 = kc * 32;
        const __bf16* ap = Arow + k0 + halfSel * 16;
        v16bf a = *(const v16bf*)(ap);
        v16bf b = *(const v16bf*)(WTn + k0 + halfSel * 16);
        acc = __builtin_amdgcn_wmma_f32_16x16x32_bf16(false, a, false, b,
                                                      (short)0, acc, false, false);
    }
    return acc;
}

__device__ __forceinline__ float sigmoidf_fast(float x) {
    return 1.0f / (1.0f + __expf(-x));
}

// ---------------------------- persistent kernel ----------------------------

__global__ __launch_bounds__(BLOCK, 1) void k_gru_scan(
    const __bf16* __restrict__ ubf,
    const __bf16* __restrict__ WTz, const __bf16* __restrict__ WTr,
    const __bf16* __restrict__ WTh, const __bf16* __restrict__ WTy,
    const float* __restrict__ bz, const float* __restrict__ br,
    const float* __restrict__ bh, const float* __restrict__ by,
    float* __restrict__ h, __bf16* __restrict__ hbf,
    float* __restrict__ zbuf, __bf16* __restrict__ rhbf,
    float* __restrict__ logits, float* __restrict__ out,
    unsigned* __restrict__ cnt, unsigned* __restrict__ gen) {

    const int tid     = threadIdx.x;
    const int lane    = tid & 31;
    const int wave    = tid >> 5;
    const int gw      = blockIdx.x * (BLOCK / 32) + wave;   // 0..511
    const int halfSel = (lane >> 4) & 1;
    const int l15     = lane & 15;

    // Gate-GEMM tile: output [64 x 1024] -> 4 x 64 tiles of 16x16.
    const int tileId = gw & 255;
    const int mt = tileId >> 6;             // batch tile 0..3
    const int nt = tileId & 63;             // hidden tile 0..63
    const bool isR = (gw >= 256);
    const int mrow = mt * 16 + l15;         // batch row this lane loads A for
    const int ncol = nt * 16 + l15;         // hidden col this lane loads B for
    const int em   = mt * 16 + halfSel * 8; // epilogue row base (C layout)
    const int en   = nt * 16 + l15;         // epilogue col

    // y-GEMM tile (only gw < 16): output [64 x 64] -> 4 x 4 tiles.
    const int mt3 = gw >> 2, nt3 = gw & 3;
    const int em3 = mt3 * 16 + halfSel * 8;
    const int en3 = nt3 * 16 + l15;

    const __bf16* Ah  = hbf  + (size_t)mrow * N_H;
    const __bf16* Arh = rhbf + (size_t)mrow * N_H;
    const __bf16* A3  = hbf  + (size_t)(mt3 * 16 + l15) * N_H;
    const __bf16* Wg  = (isR ? WTr : WTz) + (size_t)ncol * KCAT;
    const __bf16* Wg2 = WTh + (size_t)ncol * KCAT;
    const __bf16* Wy  = WTy + (size_t)en3 * N_H;

    for (int t = 0; t < TSTEPS; ++t) {
        const __bf16* Urow = ubf + ((size_t)mrow * TSTEPS + t) * N_IN;

        // -------- Phase 1: z and r (all 512 waves, one tile each) ----------
        {
            v8f acc = gemm_tile_kcat(Ah, Urow, Wg, halfSel);
            if (!isR) {
#pragma unroll
                for (int j = 0; j < 8; ++j) {
                    int idx = (em + j) * N_H + en;
                    zbuf[idx] = sigmoidf_fast(acc[j] + bz[en]);
                }
            } else {
#pragma unroll
                for (int j = 0; j < 8; ++j) {
                    int idx = (em + j) * N_H + en;
                    float r = sigmoidf_fast(acc[j] + br[en]);
                    rhbf[idx] = (__bf16)(r * h[idx]);
                }
            }
        }
        grid_barrier(cnt, gen);

        // -------- Phase 2: h_tilde + fused h update (waves 0..255) ---------
        if (!isR) {
            v8f acc = gemm_tile_kcat(Arh, Urow, Wg2, halfSel);
#pragma unroll
            for (int j = 0; j < 8; ++j) {
                int idx = (em + j) * N_H + en;
                float ht = tanhf(acc[j] + bh[en]);
                float z  = zbuf[idx];
                float hn = (1.0f - z) * h[idx] + z * ht;
                h[idx]   = hn;
                hbf[idx] = (__bf16)hn;
            }
        }
        grid_barrier(cnt, gen);

        // -------- Phase 3: y logits = h_new @ W_hy + b_y (waves 0..15) -----
        if (gw < 16) {
            v8f acc = gemm_tile_h(A3, Wy, halfSel);
#pragma unroll
            for (int j = 0; j < 8; ++j) {
                logits[(em3 + j) * N_OUT + en3] = acc[j] + by[en3];
            }
        }
        grid_barrier(cnt, gen);

        // -------- Phase 4: softmax rows -> d_out (WG0, 64 threads) ---------
        if (blockIdx.x == 0 && tid < BATCH) {
            const float* lr = logits + tid * N_OUT;
            float mx = -3.402823466e38f;
#pragma unroll 8
            for (int j = 0; j < N_OUT; ++j) mx = fmaxf(mx, lr[j]);
            float s = 0.0f;
#pragma unroll 8
            for (int j = 0; j < N_OUT; ++j) s += __expf(lr[j] - mx);
            float inv = 1.0f / s;
            float* orow = out + ((size_t)tid * TSTEPS + t) * N_OUT;
#pragma unroll 8
            for (int j = 0; j < N_OUT; ++j) orow[j] = __expf(lr[j] - mx) * inv;
        }
        // No barrier needed here: next phase 1 touches neither logits nor out,
        // and the next write to logits is two barriers away.
    }
}

// ------------------------------- host launch -------------------------------

extern "C" void kernel_launch(void* const* d_in, const int* in_sizes, int n_in,
                              void* d_out, int out_size, void* d_ws, size_t ws_size,
                              hipStream_t stream) {
    const float* u   = (const float*)d_in[0];
    const float* Wuz = (const float*)d_in[1];
    const float* Wur = (const float*)d_in[2];
    const float* Wuh = (const float*)d_in[3];
    const float* Whz = (const float*)d_in[4];
    const float* Whr = (const float*)d_in[5];
    const float* Whh = (const float*)d_in[6];
    const float* Why = (const float*)d_in[7];
    const float* bz  = (const float*)d_in[8];
    const float* br  = (const float*)d_in[9];
    const float* bh  = (const float*)d_in[10];
    const float* by  = (const float*)d_in[11];
    const float* h0  = (const float*)d_in[12];
    float* out = (float*)d_out;

    char* ws = (char*)d_ws;
    size_t o = 0;
    auto take = [&](size_t bytes) -> char* {
        char* p = ws + o;
        o += (bytes + 255) & ~(size_t)255;
        return p;
    };
    __bf16* WTz  = (__bf16*)take((size_t)N_H * KCAT * 2);
    __bf16* WTr  = (__bf16*)take((size_t)N_H * KCAT * 2);
    __bf16* WTh  = (__bf16*)take((size_t)N_H * KCAT * 2);
    __bf16* WTy  = (__bf16*)take((size_t)N_OUT * N_H * 2);
    __bf16* ubf  = (__bf16*)take((size_t)BATCH * TSTEPS * N_IN * 2);
    float*  h    = (float*) take((size_t)BATCH * N_H * 4);
    __bf16* hbf  = (__bf16*)take((size_t)BATCH * N_H * 2);
    float*  zbuf = (float*) take((size_t)BATCH * N_H * 4);
    __bf16* rhbf = (__bf16*)take((size_t)BATCH * N_H * 2);
    float*  logi = (float*) take((size_t)BATCH * N_OUT * 4);
    unsigned* bar = (unsigned*)take(256);

    const int nu = BATCH * TSTEPS * N_IN;
    k_cvt_u<<<(nu + 255) / 256, 256, 0, stream>>>(u, ubf, nu);

    const int ng = N_H * KCAT;
    k_wt_gate<<<(ng + 255) / 256, 256, 0, stream>>>(Whz, Wuz, WTz);
    k_wt_gate<<<(ng + 255) / 256, 256, 0, stream>>>(Whr, Wur, WTr);
    k_wt_gate<<<(ng + 255) / 256, 256, 0, stream>>>(Whh, Wuh, WTh);

    const int ny = N_OUT * N_H;
    k_wt_y<<<(ny + 255) / 256, 256, 0, stream>>>(Why, WTy);

    const int nh = BATCH * N_H;
    k_init_h<<<(nh + 255) / 256, 256, 0, stream>>>(h0, h, hbf, bar);

    k_gru_scan<<<NWG, BLOCK, 0, stream>>>(
        ubf, WTz, WTr, WTh, WTy, bz, br, bh, by,
        h, hbf, zbuf, rhbf, logi, out, bar, bar + 1);
}
